// ScoringLayer_17901423690383
// MI455X (gfx1250) — matrix operand order repaired and checked by reading, not compiled, not run on hardware
//
#include <hip/hip_runtime.h>
#include <hip/hip_bf16.h>

typedef __attribute__((ext_vector_type(16))) _Float16 v16h;
typedef __attribute__((ext_vector_type(8)))  _Float16 v8h;
typedef __attribute__((ext_vector_type(4)))  _Float16 v4h;
typedef __attribute__((ext_vector_type(8)))  float    v8f;

#define BATCH 2048
#define VOCAB 256
#define DDIM  16384
#define MBLK  32
#define KSTEP 64            // K elements staged per round
#define ROWP  72            // f16 per LDS row: 64 data + 8 pad (144B stride, 16B-divisible)
#define NROUND (DDIM / KSTEP)  // 256

__device__ __forceinline__ v8f wmma_f16(v16h a, v16h b, v8f c) {
  return __builtin_amdgcn_wmma_f32_16x16x32_f16(false, a, false, b, (short)0, c,
                                                false, false);
}

__device__ __forceinline__ v16h load_afrag(const _Float16* ap) {
  // 16-bit A 16x32 layout: lane holds K = kk+hi*8..+7 (v0-3) and kk+16+hi*8..+7 (v4-7)
  v8h lo = *(const v8h*)(ap);
  v8h hh = *(const v8h*)(ap + 16);
  v16h a;
#pragma unroll
  for (int i = 0; i < 8; ++i) { a[i] = lo[i]; a[8 + i] = hh[i]; }
  return a;
}

// ---------------- prep: logw[v][d] = (f16) log(10 * sigmoid(raw[v][d])) ----------------
__global__ __launch_bounds__(256) void scoring_logw_kernel(const float* __restrict__ raw,
                                                           _Float16* __restrict__ logw) {
  int i = (blockIdx.x * 256 + threadIdx.x) * 4;
  float4 x = *(const float4*)(raw + i);
  v4h o;
  o[0] = (_Float16)(2.3025851f - __logf(1.0f + __expf(-x.x)));
  o[1] = (_Float16)(2.3025851f - __logf(1.0f + __expf(-x.y)));
  o[2] = (_Float16)(2.3025851f - __logf(1.0f + __expf(-x.z)));
  o[3] = (_Float16)(2.3025851f - __logf(1.0f + __expf(-x.w)));
  *(v4h*)(logw + i) = o;
}

// ---------------- main GEMM + fused n_active + exp epilogue ----------------
// A  = binary (BATCH x DDIM, f32 in {0,1})
// Bt = logw   (VOCAB x DDIM, f16)
// out[b][v] = exp( (sum_d A[b][d]*Bt[v][d]) / max(rowsum(A[b]), 1) )
__global__ __launch_bounds__(256) void scoring_gemm_kernel(const float* __restrict__ A,
                                                           const _Float16* __restrict__ Bt,
                                                           float* __restrict__ out) {
  __shared__ __align__(16) _Float16 sA[2][MBLK * ROWP];  // double-buffered A stage
  __shared__ float sInv[MBLK];

  const int t    = threadIdx.x;
  const int wave = t >> 5;
  const int lane = t & 31;
  const int m_base = blockIdx.x * MBLK;

  // each wave owns a unique 32-column slice (2 N-tiles) and BOTH 16-row M tiles
  const int nb = wave * 32;

  const int lrow = t >> 3;   // staging row 0..31
  const int lcg  = t & 7;    // staging group: 8 floats along K

  const int r  = lane & 15;  // fragment row/col within a 16x16 tile
  const int hi = lane >> 4;  // K-half selector

  v8f acc00 = {}, acc01 = {}, acc10 = {}, acc11 = {};
  float nsum = 0.0f;

  const float*    aRow  = A  + (size_t)(m_base + lrow) * DDIM + lcg * 8;
  const _Float16* bBase = Bt + (size_t)(nb + r) * DDIM + hi * 8;

  // ---- prologue: stage round 0 into buffer 0 ----
  {
    float4 x0 = *(const float4*)(aRow);
    float4 x1 = *(const float4*)(aRow + 4);
    nsum += x0.x + x0.y + x0.z + x0.w + x1.x + x1.y + x1.z + x1.w;
    v8h h;
    h[0] = (_Float16)x0.x; h[1] = (_Float16)x0.y; h[2] = (_Float16)x0.z; h[3] = (_Float16)x0.w;
    h[4] = (_Float16)x1.x; h[5] = (_Float16)x1.y; h[6] = (_Float16)x1.z; h[7] = (_Float16)x1.w;
    *(v8h*)(sA[0] + lrow * ROWP + lcg * 8) = h;
  }
  __syncthreads();

  for (int rd = 0; rd < NROUND - 1; ++rd) {
    const int cur = rd & 1;
    const int k0  = rd * KSTEP;

    // issue global loads for the NEXT round first (latency hiding)
    float4 x0 = *(const float4*)(aRow + k0 + KSTEP);
    float4 x1 = *(const float4*)(aRow + k0 + KSTEP + 4);

    // prefetch A two rounds ahead (global_prefetch_b8)
    {
      int pf = (rd + 2 < NROUND) ? (rd + 2) : (NROUND - 1);
      __builtin_prefetch(aRow + pf * KSTEP, 0, 1);
    }

    // ---- compute from sA[cur]: 2 K-substeps x (2M x 2N) WMMAs ----
    const _Float16* base = sA[cur];
#pragma unroll
    for (int kk = 0; kk < KSTEP; kk += 32) {
      v16h a0 = load_afrag(base + r * ROWP + kk + hi * 8);
      v16h a1 = load_afrag(base + (16 + r) * ROWP + kk + hi * 8);
      const _Float16* bp = bBase + k0 + kk;
      v16h b0 = load_afrag(bp);
      v16h b1 = load_afrag(bp + (size_t)16 * DDIM);
      acc00 = wmma_f16(a0, b0, acc00);
      acc01 = wmma_f16(a0, b1, acc01);
      acc10 = wmma_f16(a1, b0, acc10);
      acc11 = wmma_f16(a1, b1, acc11);
    }

    // ---- stage next round into the other buffer ----
    nsum += x0.x + x0.y + x0.z + x0.w + x1.x + x1.y + x1.z + x1.w;
    v8h h;
    h[0] = (_Float16)x0.x; h[1] = (_Float16)x0.y; h[2] = (_Float16)x0.z; h[3] = (_Float16)x0.w;
    h[4] = (_Float16)x1.x; h[5] = (_Float16)x1.y; h[6] = (_Float16)x1.z; h[7] = (_Float16)x1.w;
    *(v8h*)(sA[1 - cur] + lrow * ROWP + lcg * 8) = h;
    __syncthreads();
  }

  // ---- last round: compute only ----
  {
    const int rd  = NROUND - 1;
    const int cur = rd & 1;
    const int k0  = rd * KSTEP;
    const _Float16* base = sA[cur];
#pragma unroll
    for (int kk = 0; kk < KSTEP; kk += 32) {
      v16h a0 = load_afrag(base + r * ROWP + kk + hi * 8);
      v16h a1 = load_afrag(base + (16 + r) * ROWP + kk + hi * 8);
      const _Float16* bp = bBase + k0 + kk;
      v16h b0 = load_afrag(bp);
      v16h b1 = load_afrag(bp + (size_t)16 * DDIM);
      acc00 = wmma_f16(a0, b0, acc00);
      acc01 = wmma_f16(a0, b1, acc01);
      acc10 = wmma_f16(a1, b0, acc10);
      acc11 = wmma_f16(a1, b1, acc11);
    }
  }

  // ---- n_active: 8 consecutive lanes share one staged row ----
  nsum += __shfl_xor(nsum, 1, 32);
  nsum += __shfl_xor(nsum, 2, 32);
  nsum += __shfl_xor(nsum, 4, 32);
  if ((lane & 7) == 0) sInv[lrow] = 1.0f / fmaxf(nsum, 1.0f);
  __syncthreads();

  // ---- epilogue: C layout — VGPR j of lane L holds (M = (L>>4)*8 + j, N = L&15) ----
#pragma unroll
  for (int j = 0; j < 8; ++j) {
    int lr0 = hi * 8 + j;        // M tile 0 (rows 0-15)
    int lr1 = 16 + lr0;          // M tile 1 (rows 16-31)
    float s0 = sInv[lr0];
    float s1 = sInv[lr1];
    float* op0 = out + (size_t)(m_base + lr0) * VOCAB + nb + r;
    float* op1 = out + (size_t)(m_base + lr1) * VOCAB + nb + r;
    op0[0]  = __expf(acc00[j] * s0);
    op0[16] = __expf(acc01[j] * s0);
    op1[0]  = __expf(acc10[j] * s1);
    op1[16] = __expf(acc11[j] * s1);
  }
}

extern "C" void kernel_launch(void* const* d_in, const int* in_sizes, int n_in,
                              void* d_out, int out_size, void* d_ws, size_t ws_size,
                              hipStream_t stream) {
  const float* binary = (const float*)d_in[0];   // (BATCH, DDIM) f32 in {0,1}
  const float* raw    = (const float*)d_in[1];   // (VOCAB, DDIM) f32
  float* out = (float*)d_out;                    // (BATCH, VOCAB) f32
  _Float16* logw = (_Float16*)d_ws;              // VOCAB*DDIM*2 = 8 MB scratch

  const int nlogw = VOCAB * DDIM;                            // 4,194,304
  scoring_logw_kernel<<<nlogw / (256 * 4), 256, 0, stream>>>(raw, logw);
  scoring_gemm_kernel<<<BATCH / MBLK, 256, 0, stream>>>(binary, logw, out);
}
